// TreeCRF_3118146257274
// MI455X (gfx1250) — matrix operand order repaired and checked by reading, not compiled, not run on hardware
//
#include <hip/hip_runtime.h>

// Biaffine attention for MI455X (gfx1250), wave32, bf16 WMMA with f32 accumulate.
// out[b,n,l,m] = (h[b,l]·U[n]·c[b,m] + Wd[n]·h[b,l] + We[n]·c[b,m] + bias[n])
//                * mask[b,l]*mask[b,m],  diag -> -inf
#define BB 32
#define LL 512
#define DD 512
#define NN 8
#define TM 64          // l-rows per workgroup
#define RS 520         // LDS row stride (halves) for 512-wide bf16 tiles (+8 pad)
#define RSU 40         // LDS row stride (halves) for transposed U chunk (32-wide +8 pad)

#if defined(__has_builtin)
#  if __has_builtin(__builtin_amdgcn_sched_barrier)
#    define SCHED_FENCE() __builtin_amdgcn_sched_barrier(0)
#  endif
#endif
#ifndef SCHED_FENCE
#  define SCHED_FENCE()
#endif

typedef unsigned short u16;
typedef unsigned int   u32;
typedef __bf16 v16bf __attribute__((ext_vector_type(16)));
typedef float  v8f   __attribute__((ext_vector_type(8)));

struct alignas(16) F4 { float x, y, z, w; };
struct alignas(16) Q4 { u32 a, b, c, d; };
union  FragU { Q4 q[2]; v16bf v; };

__device__ __forceinline__ u16 f2bf(float x) {   // round-to-nearest-even f32 -> bf16
  u32 u = __float_as_uint(x);
  return (u16)((u + 0x7FFFu + ((u >> 16) & 1u)) >> 16);
}
__device__ __forceinline__ u32 pack2(float lo, float hi) {
  return (u32)f2bf(lo) | ((u32)f2bf(hi) << 16);
}

// A-matrix 16x32 bf16 fragment (ISA 7.12.2): lane m<16 row M=m, halves 0..7 = K(k0+hh*8+i),
// halves 8..15 = K(k0+16+hh*8+i); hh = lane/16.
__device__ __forceinline__ v16bf frag_a(const u16* base, int stride, int row0, int k0, int lane) {
  int m = lane & 15, hh = lane >> 4;
  const u16* p = base + (row0 + m) * stride + k0 + hh * 8;
  FragU f;
  f.q[0] = *(const Q4*)(p);
  f.q[1] = *(const Q4*)(p + 16);
  return f.v;
}
// B-matrix 32x16 bf16 fragment: lane n<16 col N=n, halves 0..7 = K(k0+hh*8+i), halves 8..15 = K(k0+16+hh*8+i).
// Source staged K-contiguous: bt[col][k].
__device__ __forceinline__ v16bf frag_b(const u16* base, int stride, int col0, int k0, int lane) {
  int nn = lane & 15, hh = lane >> 4;
  const u16* p = base + (col0 + nn) * stride + k0 + hh * 8;
  FragU f;
  f.q[0] = *(const Q4*)(p);
  f.q[1] = *(const Q4*)(p + 16);
  return f.v;
}

__global__ __launch_bounds__(256)
void biaffine_wmma_kernel(const float* __restrict__ h, const float* __restrict__ c,
                          const float* __restrict__ mask, const float* __restrict__ U,
                          const float* __restrict__ Wd, const float* __restrict__ We,
                          const float* __restrict__ bias, float* __restrict__ out) {
  __shared__ u16 sm0[TM * RS];            // stage A: H tile (bf16). stage B: T tile (bf16), aliased.
  __shared__ u16 sm1[TM * RS];            // stage A: U chunk transposed (uses 512*RSU). stage B: C tile.
  __shared__ float spart[256];
  __shared__ float slin_l[TM], slin_m[TM], smask_l[TM], smask_m[TM];

  const int tid  = threadIdx.x;
  const int lane = tid & 31;
  const int w    = tid >> 5;              // 8 waves
  const int lblk = blockIdx.x, n = blockIdx.y, b = blockIdx.z;
  const int l0   = lblk * TM;

  const float* hB = h + ((size_t)b * LL + l0) * DD;
  const float* cB = c + (size_t)b * LL * DD;
  const float* Un = U + (size_t)n * DD * DD;
  const float  bn = bias[n];

  // ---- stage H tile f32->bf16, compute lin_l = Wd[n]·h row, cache mask_l ----
  for (int it = 0; it < 32; ++it) {
    int idx = it * 256 + tid;             // float4 index over 64x128
    int row = idx >> 7;
    int c4  = (idx & 127) << 2;
    F4 v = *(const F4*)(hB + row * DD + c4);
    u32* d = (u32*)(sm0 + row * RS + c4);
    d[0] = pack2(v.x, v.y);
    d[1] = pack2(v.z, v.w);
  }
  {
    int r = tid >> 2, p = tid & 3;        // 4 partials per row
    const float* hr = hB + (size_t)r * DD + p * 128;
    const float* wd = Wd + n * DD + p * 128;
    float s = 0.f;
    for (int e = 0; e < 128; ++e) s += wd[e] * hr[e];
    spart[tid] = s;
  }
  if (tid < TM) smask_l[tid] = mask[b * LL + l0 + tid];
  __syncthreads();
  if (tid < TM) slin_l[tid] = spart[4*tid] + spart[4*tid+1] + spart[4*tid+2] + spart[4*tid+3];

  // ---- stage A: T(64x512) = Htile @ U[n]; wave grid 4(rows) x 2(256-wide e strips) ----
  const int r0     = (w & 3) * 16;
  const int e0base = (w >> 2) * 256;
  v8f zero = {};
  v8f accA[16];
#pragma unroll
  for (int et = 0; et < 16; ++et) accA[et] = zero;

  for (int kc = 0; kc < 16; ++kc) {       // K chunks of 32 over d
    __syncthreads();                      // everyone done reading previous U chunk
    const float* Up = Un + (size_t)(kc * 32) * DD;
    // prefetch next U chunk into L2 (global_prefetch_b8); 256B stride covers 64KB
    if (kc < 15)
      __builtin_prefetch((const char*)(Up + (size_t)32 * DD) + tid * 256, 0, 1);
    for (int it = 0; it < 16; ++it) {     // stage transposed: sm1[e][d_local]
      int idx = it * 256 + tid;           // float4 index over 32x128
      int dl  = idx >> 7;
      int e4  = (idx & 127) << 2;
      F4 v = *(const F4*)(Up + dl * DD + e4);
      sm1[(e4 + 0) * RSU + dl] = f2bf(v.x);
      sm1[(e4 + 1) * RSU + dl] = f2bf(v.y);
      sm1[(e4 + 2) * RSU + dl] = f2bf(v.z);
      sm1[(e4 + 3) * RSU + dl] = f2bf(v.w);
    }
    __syncthreads();
    v16bf af  = frag_a(sm0, RS, r0, kc * 32, lane);
    // depth-2 software pipeline of B fragments, pinned with sched fences
    v16bf bq0 = frag_b(sm1, RSU, e0base + 0,  0, lane);
    v16bf bq1 = frag_b(sm1, RSU, e0base + 16, 0, lane);
#pragma unroll
    for (int et = 0; et < 16; ++et) {
      v16bf bnx = frag_b(sm1, RSU, e0base + (((et + 2) & 15) << 4), 0, lane);
      SCHED_FENCE();                      // keep the lookahead loads above this wmma
      accA[et] = __builtin_amdgcn_wmma_f32_16x16x32_bf16(
          false, af, false, bq0, (short)0, accA[et], false, false);
      bq0 = bq1; bq1 = bnx;
    }
  }

  // ---- spill T as bf16 over the dead H region (barrier first) ----
  __syncthreads();
  {
    int nn = lane & 15, hh = lane >> 4;   // C/D layout: vgpr i, lane j -> (M=i+8*(j/16), N=j%16)
#pragma unroll
    for (int et = 0; et < 16; ++et) {
      int col = e0base + et * 16 + nn;
#pragma unroll
      for (int i = 0; i < 8; ++i)
        sm0[(r0 + i + hh * 8) * RS + col] = f2bf(accA[et][i]);
    }
  }

  // ---- stage B: out(64x64 per m-chunk) = T @ Ctile^T, fused epilogue ----
  const int c0 = (w >> 2) * 32;           // wave grid 4(rows) x 2(32-wide m strips), 2 tiles each
  const size_t outBase = (size_t)(b * NN + n) * LL * LL;
  for (int mc = 0; mc < 8; ++mc) {
    const int m0 = mc * TM;
    __syncthreads();                      // T writes + previous epilogue done
    const float* cT = cB + (size_t)m0 * DD;
    // prefetch next C tile into L2 (128KB; two 256B-strided lines per thread)
    if (mc < 7) {
      const char* pf = (const char*)(cT + (size_t)TM * DD);
      __builtin_prefetch(pf + tid * 512, 0, 1);
      __builtin_prefetch(pf + tid * 512 + 256, 0, 1);
    }
    for (int it = 0; it < 32; ++it) {     // stage C tile: sm1[m][e] bf16
      int idx = it * 256 + tid;
      int row = idx >> 7;
      int c4  = (idx & 127) << 2;
      F4 v = *(const F4*)(cT + row * DD + c4);
      u32* d = (u32*)(sm1 + row * RS + c4);
      d[0] = pack2(v.x, v.y);
      d[1] = pack2(v.z, v.w);
    }
    {
      int r = tid >> 2, p = tid & 3;
      const float* cr = cT + (size_t)r * DD + p * 128;
      const float* we = We + n * DD + p * 128;
      float s = 0.f;
      for (int e = 0; e < 128; ++e) s += we[e] * cr[e];
      spart[tid] = s;
    }
    if (tid < TM) smask_m[tid] = mask[b * LL + m0 + tid];
    __syncthreads();
    if (tid < TM) slin_m[tid] = spart[4*tid] + spart[4*tid+1] + spart[4*tid+2] + spart[4*tid+3];
    __syncthreads();

    v8f acc0 = zero, acc1 = zero;
    // depth-1 pipeline: fragments for kc+1 issued (and fenced) before the wmmas of kc
    v16bf af = frag_a(sm0, RS, r0, 0, lane);
    v16bf b0 = frag_b(sm1, RS, c0, 0, lane);
    v16bf b1 = frag_b(sm1, RS, c0 + 16, 0, lane);
    for (int kc = 0; kc < 16; ++kc) {
      int kn = ((kc + 1) & 15) * 32;
      v16bf afn = frag_a(sm0, RS, r0, kn, lane);
      v16bf b0n = frag_b(sm1, RS, c0, kn, lane);
      v16bf b1n = frag_b(sm1, RS, c0 + 16, kn, lane);
      SCHED_FENCE();                      // keep kc+1 loads above the kc wmmas
      acc0 = __builtin_amdgcn_wmma_f32_16x16x32_bf16(false, af, false, b0, (short)0, acc0, false, false);
      acc1 = __builtin_amdgcn_wmma_f32_16x16x32_bf16(false, af, false, b1, (short)0, acc1, false, false);
      af = afn; b0 = b0n; b1 = b1n;
    }

    // epilogue: linear terms + bias, masking, diagonal -inf, non-temporal f32 store
    int nn = lane & 15, hh = lane >> 4;
#pragma unroll
    for (int tt = 0; tt < 2; ++tt) {
      int cm = c0 + tt * 16 + nn;
      int gm = m0 + cm;
      float colAdd = slin_m[cm] + bn;
      float mm     = smask_m[cm];
#pragma unroll
      for (int i = 0; i < 8; ++i) {
        int r  = r0 + i + hh * 8;
        int gl = l0 + r;
        float v = ((tt == 0 ? acc0[i] : acc1[i]) + slin_l[r] + colAdd) * (smask_l[r] * mm);
        if (gl == gm) v = -__builtin_inff();
        __builtin_nontemporal_store(v, &out[outBase + (size_t)gl * LL + gm]);
      }
    }
  }
}

extern "C" void kernel_launch(void* const* d_in, const int* in_sizes, int n_in,
                              void* d_out, int out_size, void* d_ws, size_t ws_size,
                              hipStream_t stream) {
  const float* h    = (const float*)d_in[0];
  const float* c    = (const float*)d_in[1];
  const float* mask = (const float*)d_in[2];
  const float* U    = (const float*)d_in[3];
  const float* Wd   = (const float*)d_in[4];
  const float* We   = (const float*)d_in[5];
  const float* bias = (const float*)d_in[6];
  float* out = (float*)d_out;
  (void)in_sizes; (void)n_in; (void)out_size; (void)d_ws; (void)ws_size;

  dim3 grid(LL / TM, NN, BB);   // (8, 8, 32) = 2048 workgroups of 8 wave32s
  biaffine_wmma_kernel<<<grid, dim3(256), 0, stream>>>(h, c, mask, U, Wd, We, bias, out);
}
